// FunctionalMamba_14405320311411
// MI455X (gfx1250) — matrix operand order repaired and checked by reading, not compiled, not run on hardware
//
#include <hip/hip_runtime.h>
#include <hip/hip_bf16.h>

// ---------------------------------------------------------------- constants
#define B_   2
#define L_   2048
#define DM_  1024
#define DI_  2048
#define N_   16
#define R_   64
#define KC_  4          // conv kernel width
#define M_   (B_ * L_)  // 4096 rows
#define EPS_ 1e-5f

typedef __attribute__((ext_vector_type(16))) __bf16        v16bf;
typedef __attribute__((ext_vector_type(8)))  float         v8f;
typedef __attribute__((ext_vector_type(8)))  unsigned int  v8u;

// ================================================================ LayerNorm
__global__ __launch_bounds__(256)
void ln_kernel(const float* __restrict__ x, const float* __restrict__ w,
               const float* __restrict__ b, float* __restrict__ out) {
    __shared__ float s_sum[8], s_sq[8];
    const int row = blockIdx.x;
    const float* xr = x + (size_t)row * DM_;
    float v[4], s = 0.f, sq = 0.f;
#pragma unroll
    for (int i = 0; i < 4; ++i) {
        v[i] = xr[threadIdx.x + i * 256];
        s += v[i]; sq += v[i] * v[i];
    }
#pragma unroll
    for (int off = 16; off; off >>= 1) {
        s  += __shfl_xor(s,  off, 32);
        sq += __shfl_xor(sq, off, 32);
    }
    const int lane = threadIdx.x & 31, wv = threadIdx.x >> 5;
    if (lane == 0) { s_sum[wv] = s; s_sq[wv] = sq; }
    __syncthreads();
    if (wv == 0) {
        s  = (lane < 8) ? s_sum[lane] : 0.f;
        sq = (lane < 8) ? s_sq[lane]  : 0.f;
#pragma unroll
        for (int off = 4; off; off >>= 1) {
            s  += __shfl_xor(s,  off, 32);
            sq += __shfl_xor(sq, off, 32);
        }
        if (lane == 0) { s_sum[0] = s; s_sq[0] = sq; }
    }
    __syncthreads();
    const float mu  = s_sum[0] * (1.f / DM_);
    const float var = s_sq[0] * (1.f / DM_) - mu * mu;
    const float r   = rsqrtf(var + EPS_);
    float* outr = out + (size_t)row * DM_;
#pragma unroll
    for (int i = 0; i < 4; ++i) {
        const int c = threadIdx.x + i * 256;
        outr[c] = (v[i] - mu) * r * w[c] + b[c];
    }
}

// ============================================= GEMM: OUT = X(MxK) * W(NxK)^T
// bf16 hi/lo error-compensated WMMA (≈fp32 accuracy, f32 accumulate).
// Tiles in LDS as packed bf16-pair dwords (hi/lo planes), pitch 20 dwords:
// rows 16B-aligned (-> ds_load_b128 fragments) and 20*r mod 64 puts the 16
// lanes' b128 reads on disjoint bank quartets (conflict-free).
// Register-staged software pipeline: global loads for tile t+1 issue BEFORE
// the 6 WMMAs of tile t; cvt+LDS-store happens after, so s_wait_loadcnt is
// covered by matrix work. One barrier per K-iteration.
// Block: 256 thr (8 waves). Tile: 128(M) x 32(N). Wave w owns rows w*16..+15.
// EPI==1: softplus(acc + bias[n])  (dt_proj epilogue).
template <int EPI>
__global__ __launch_bounds__(256)
void gemm_xwT(const float* __restrict__ X, const float* __restrict__ W,
              const float* __restrict__ bias, float* __restrict__ OUT,
              int Ktot, int ldx, int Ntot) {
    __shared__ unsigned int sXhi[2][128][20], sXlo[2][128][20];
    __shared__ unsigned int sWhi[2][32][20],  sWlo[2][32][20];

    const int tid  = threadIdx.x;
    const int w    = tid >> 5;
    const int lane = tid & 31;
    const int lm   = lane & 15;     // m (A) / n (B) within 16x16 subtile
    const int kh   = lane >> 4;     // K-half selector

    const int n0 = blockIdx.x * 32;
    const int m0 = blockIdx.y * 128;

    // per-thread staging coordinates for cooperative tile loads
    const int xr_[4] = { (tid + 0) >> 3, (tid + 256) >> 3,
                         (tid + 512) >> 3, (tid + 768) >> 3 };
    const int xc_ = (tid & 7) * 4;            // float col within 32-wide tile
    const int wr_ = tid >> 3;                 // W row (0..31)

    v8f acc[2] = {};

    auto cvt2 = [](float f0, float f1, unsigned int& hi, unsigned int& lo) {
        const __bf16 h0 = (__bf16)f0, h1 = (__bf16)f1;
        const __bf16 l0 = (__bf16)(f0 - (float)h0);
        const __bf16 l1 = (__bf16)(f1 - (float)h1);
        hi = (unsigned int)__builtin_bit_cast(unsigned short, h0)
           | ((unsigned int)__builtin_bit_cast(unsigned short, h1) << 16);
        lo = (unsigned int)__builtin_bit_cast(unsigned short, l0)
           | ((unsigned int)__builtin_bit_cast(unsigned short, l1) << 16);
    };

    // stage 1: issue global loads into registers (no LDS, no waits forced)
    auto g_load = [&](int k0, float4 (&rx)[4], float4& rw) {
#pragma unroll
        for (int j = 0; j < 4; ++j)
            rx[j] = *(const float4*)&X[(size_t)(m0 + xr_[j]) * ldx + k0 + xc_];
        rw = *(const float4*)&W[(size_t)(n0 + wr_) * Ktot + k0 + xc_];
    };
    // stage 2: convert hi/lo and store packed pairs to LDS
    auto cvt_store = [&](int buf, const float4 (&rx)[4], const float4& rw) {
        const int cw = xc_ >> 1;              // packed-pair word index
#pragma unroll
        for (int j = 0; j < 4; ++j) {
            unsigned int h0, l0, h1, l1;
            cvt2(rx[j].x, rx[j].y, h0, l0);
            cvt2(rx[j].z, rx[j].w, h1, l1);
            sXhi[buf][xr_[j]][cw + 0] = h0;  sXhi[buf][xr_[j]][cw + 1] = h1;
            sXlo[buf][xr_[j]][cw + 0] = l0;  sXlo[buf][xr_[j]][cw + 1] = l1;
        }
        unsigned int h0, l0, h1, l1;
        cvt2(rw.x, rw.y, h0, l0);
        cvt2(rw.z, rw.w, h1, l1);
        sWhi[buf][wr_][cw + 0] = h0;  sWhi[buf][wr_][cw + 1] = h1;
        sWlo[buf][wr_][cw + 0] = l0;  sWlo[buf][wr_][cw + 1] = l1;
    };

    float4 rx[4]; float4 rw;
    g_load(0, rx, rw);
    cvt_store(0, rx, rw);

    int buf = 0;
    for (int k0 = 0; k0 < Ktot; k0 += 32) {
        __syncthreads();
        const bool nxt = (k0 + 32 < Ktot);
        if (nxt) g_load(k0 + 32, rx, rw);     // in flight during WMMAs

        // ---- A fragments: word = (v<4 ? v : v+4) + 4*kh  (2x ds_load_b128)
        v8u au_hi, au_lo;
#pragma unroll
        for (int v = 0; v < 8; ++v) {
            const int idx = ((v < 4) ? v : (v + 4)) + 4 * kh;
            au_hi[v] = sXhi[buf][w * 16 + lm][idx];
            au_lo[v] = sXlo[buf][w * 16 + lm][idx];
        }
        const v16bf a_hi = __builtin_bit_cast(v16bf, au_hi);
        const v16bf a_lo = __builtin_bit_cast(v16bf, au_lo);

#pragma unroll
        for (int ns = 0; ns < 2; ++ns) {
            // ---- B fragments: word = (j<4 ? j : j+4) + 4*kh  (2x ds_load_b128)
            v8u bu_hi, bu_lo;
#pragma unroll
            for (int j = 0; j < 8; ++j) {
                bu_hi[j] = sWhi[buf][ns * 16 + lm][((j < 4) ? j : (j + 4)) + 4 * kh];
                bu_lo[j] = sWlo[buf][ns * 16 + lm][((j < 4) ? j : (j + 4)) + 4 * kh];
            }
            const v16bf b_hi = __builtin_bit_cast(v16bf, bu_hi);
            const v16bf b_lo = __builtin_bit_cast(v16bf, bu_lo);

            acc[ns] = __builtin_amdgcn_wmma_f32_16x16x32_bf16(
                false, a_lo, false, b_hi, (short)0, acc[ns], false, false);
            acc[ns] = __builtin_amdgcn_wmma_f32_16x16x32_bf16(
                false, a_hi, false, b_lo, (short)0, acc[ns], false, false);
            acc[ns] = __builtin_amdgcn_wmma_f32_16x16x32_bf16(
                false, a_hi, false, b_hi, (short)0, acc[ns], false, false);
        }

        if (nxt) cvt_store(buf ^ 1, rx, rw);  // wait lands after WMMAs
        buf ^= 1;
    }

    // ---- epilogue: D layout row = r + 8*kh, col = lm
#pragma unroll
    for (int ns = 0; ns < 2; ++ns) {
        const int n = n0 + ns * 16 + lm;
#pragma unroll
        for (int r = 0; r < 8; ++r) {
            const int m = m0 + w * 16 + r + 8 * kh;
            float val = acc[ns][r];
            if (EPI == 1) {   // softplus(acc + bias)
                const float t = val + bias[n];
                val = (t > 20.f) ? t : log1pf(__expf(t));
            }
            OUT[(size_t)m * Ntot + n] = val;
        }
    }
}

// ======================================== causal depthwise conv (K=4) + SiLU
__global__ __launch_bounds__(256)
void dwconv_silu(const float* __restrict__ xz, const float* __restrict__ cw,
                 const float* __restrict__ cb, float* __restrict__ xc) {
    const int id = blockIdx.x * 256 + threadIdx.x;   // over M_*DI_
    const int d   = id & (DI_ - 1);
    const int row = id >> 11;                        // DI_=2048
    const int l   = row & (L_ - 1);
    const int bb  = row >> 11;                       // L_=2048
    float acc = cb[d];
#pragma unroll
    for (int k = 0; k < KC_; ++k) {
        const int ll = l + k - (KC_ - 1);
        if (ll >= 0)
            acc += xz[((size_t)(bb * L_ + ll)) * (2 * DI_) + d] * cw[d * KC_ + k];
    }
    const float sg = 1.f / (1.f + __expf(-acc));
    xc[(size_t)row * DI_ + d] = acc * sg;
}

// =========================== selective scan: half-wave (16 lanes) per channel
__global__ __launch_bounds__(256)
void scan_kernel(const float* __restrict__ dtb, const float* __restrict__ xdbl,
                 const float* __restrict__ xc, const float* __restrict__ A_log,
                 const float* __restrict__ Dp, float* __restrict__ y) {
    const int wave = threadIdx.x >> 5;
    const int lane = threadIdx.x & 31;
    const int g = lane >> 4, n = lane & 15;
    const int task = blockIdx.x * 8 + wave;          // 0..B_*DI_/2-1
    const int bb = task / (DI_ / 2);
    const int d  = (task % (DI_ / 2)) * 2 + g;

    const float An = -__expf(A_log[d * N_ + n]);
    const float Dd = Dp[d];
    float h = 0.f;

    for (int l = 0; l < L_; ++l) {
        const size_t row = (size_t)(bb * L_ + l);
        const float dtv = dtb[row * DI_ + d];
        const float xv  = xc [row * DI_ + d];
        const float Bv  = xdbl[row * (R_ + 2 * N_) + R_ + n];
        const float Cv  = xdbl[row * (R_ + 2 * N_) + R_ + N_ + n];
        h = h * __expf(dtv * An) + (dtv * xv) * Bv;
        float p = h * Cv;
        p += __shfl_xor(p, 8, 16);
        p += __shfl_xor(p, 4, 16);
        p += __shfl_xor(p, 2, 16);
        p += __shfl_xor(p, 1, 16);
        if (n == 0) y[row * DI_ + d] = p + xv * Dd;
    }
}

// ================================================== gating: y *= silu(z)
__global__ __launch_bounds__(256)
void gate_kernel(float* __restrict__ y, const float* __restrict__ xz) {
    const int id = blockIdx.x * 256 + threadIdx.x;   // over M_*DI_
    const int d   = id & (DI_ - 1);
    const int row = id >> 11;
    const float z = xz[(size_t)row * (2 * DI_) + DI_ + d];
    const float sg = 1.f / (1.f + __expf(-z));
    y[(size_t)row * DI_ + d] *= z * sg;
}

// ================================================================== launcher
extern "C" void kernel_launch(void* const* d_in, const int* in_sizes, int n_in,
                              void* d_out, int out_size, void* d_ws, size_t ws_size,
                              hipStream_t stream) {
    const float* hs        = (const float*)d_in[0];
    const float* norm_w    = (const float*)d_in[1];
    const float* norm_b    = (const float*)d_in[2];
    const float* in_proj_w = (const float*)d_in[3];   // (2*DI, DM)
    const float* conv_w    = (const float*)d_in[4];   // (DI, K)
    const float* conv_b    = (const float*)d_in[5];
    const float* x_proj_w  = (const float*)d_in[6];   // (R+2N, DI)
    const float* dt_proj_w = (const float*)d_in[7];   // (DI, R)
    const float* dt_proj_b = (const float*)d_in[8];
    const float* A_log     = (const float*)d_in[9];   // (DI, N)
    const float* D_param   = (const float*)d_in[10];
    const float* out_proj_w= (const float*)d_in[11];  // (DM, DI)

    float* out = (float*)d_out;

    // workspace partition (floats)
    float* xn   = (float*)d_ws;                       // M x DM
    float* xz   = xn   + (size_t)M_ * DM_;            // M x 2*DI
    float* xc   = xz   + (size_t)M_ * 2 * DI_;        // M x DI
    float* xdbl = xc   + (size_t)M_ * DI_;            // M x 96
    float* dt   = xdbl + (size_t)M_ * (R_ + 2 * N_);  // M x DI
    float* y    = dt   + (size_t)M_ * DI_;            // M x DI

    // 1. LayerNorm
    ln_kernel<<<M_, 256, 0, stream>>>(hs, norm_w, norm_b, xn);

    // 2. in_proj: xz = xn(4096x1024) * in_proj_w^T -> 4096x4096
    gemm_xwT<0><<<dim3((2 * DI_) / 32, M_ / 128), 256, 0, stream>>>(
        xn, in_proj_w, nullptr, xz, DM_, DM_, 2 * DI_);

    // 3. causal depthwise conv + SiLU on x half
    dwconv_silu<<<(M_ * DI_) / 256, 256, 0, stream>>>(xz, conv_w, conv_b, xc);

    // 4. x_proj: xdbl = xc(4096x2048) * x_proj_w^T -> 4096x96
    gemm_xwT<0><<<dim3((R_ + 2 * N_) / 32, M_ / 128), 256, 0, stream>>>(
        xc, x_proj_w, nullptr, xdbl, DI_, DI_, R_ + 2 * N_);

    // 5. dt = softplus(dt_lo(4096x64) * dt_proj_w^T + b) -> 4096x2048
    gemm_xwT<1><<<dim3(DI_ / 32, M_ / 128), 256, 0, stream>>>(
        xdbl, dt_proj_w, dt_proj_b, dt, R_, R_ + 2 * N_, DI_);

    // 6. selective scan (2048 half-wave channels per batch*DI pair)
    scan_kernel<<<(B_ * DI_ / 2) / 8, 256, 0, stream>>>(
        dt, xdbl, xc, A_log, D_param, y);

    // 7. gating y *= silu(z)
    gate_kernel<<<(M_ * DI_) / 256, 256, 0, stream>>>(y, xz);

    // 8. out_proj: out = y(4096x2048) * out_proj_w^T -> 4096x1024
    gemm_xwT<0><<<dim3(DM_ / 32, M_ / 128), 256, 0, stream>>>(
        y, out_proj_w, nullptr, out, DI_, DI_, DM_);

    // 9. residual (second tuple element) = hidden_states
    hipMemcpyAsync(out + (size_t)M_ * DM_, hs, sizeof(float) * (size_t)M_ * DM_,
                   hipMemcpyDeviceToDevice, stream);
}
